// JiTAttention_86698209837656
// MI455X (gfx1250) — matrix operand order repaired and checked by reading, not compiled, not run on hardware
//
#include <hip/hip_runtime.h>

// ---------------------------------------------------------------------------
// Types (ext vectors only: trivially-copyable, union-safe)
// ---------------------------------------------------------------------------
typedef __attribute__((ext_vector_type(16))) __bf16        v16bf;
typedef __attribute__((ext_vector_type(8)))  float         v8f;
typedef __attribute__((ext_vector_type(4)))  unsigned int  u32x4;
typedef __attribute__((ext_vector_type(4)))  float         f32x4;

union Frag16 { v16bf v; u32x4 q[2]; unsigned short u[16]; };
union F8     { v8f   v; float f[8]; };
union H32    { u32x4 q[4]; unsigned short u[32]; };

// ---------------------------------------------------------------------------
// CDNA5 DMA paths: TDM (TENSORcnt) and async global->LDS (ASYNCcnt)
// ---------------------------------------------------------------------------
#ifndef __has_builtin
#define __has_builtin(x) 0
#endif
#if __has_builtin(__builtin_amdgcn_global_load_async_to_lds_b128) && \
    __has_builtin(__builtin_amdgcn_s_wait_asynccnt)
#define USE_ASYNC_LDS 1
#else
#define USE_ASYNC_LDS 0
#endif
#if __has_builtin(__builtin_amdgcn_tensor_load_to_lds) && \
    __has_builtin(__builtin_amdgcn_s_wait_tensorcnt)
#define USE_TDM 1
#else
#define USE_TDM 0
#endif

#if USE_ASYNC_LDS
typedef int v4i_gcc __attribute__((vector_size(16)));   // matches builtin param
typedef __attribute__((address_space(1))) v4i_gcc* g_v4i_p;
typedef __attribute__((address_space(3))) v4i_gcc* l_v4i_p;

__device__ __forceinline__ void async_b128(const unsigned short* g, unsigned short* l) {
  // per-lane 16B copy, tracked by ASYNCcnt
  __builtin_amdgcn_global_load_async_to_lds_b128((g_v4i_p)g, (l_v4i_p)l, 0, 0);
}
#endif

#if USE_TDM
typedef unsigned int u32x4g __attribute__((vector_size(16)));
typedef int          i32x4g __attribute__((vector_size(16)));
typedef int          i32x8g __attribute__((vector_size(32)));

// One 2D-tile DMA: rowsxcols bf16 tile (cols contiguous) -> LDS, row-major.
// D# layout per CDNA5 ISA ch.8: group0 {count,lds_addr,global_addr,type=2},
// group1 {data_size=2B, tensor dims, tile dims, dim0 stride}.
__device__ __forceinline__ void tdm_load_tile(const unsigned short* gsrc,
                                              unsigned short* lds_dst,
                                              int tile_cols, int tile_rows,
                                              int row_stride_elems) {
  unsigned long long ga = (unsigned long long)(size_t)gsrc;
  unsigned int laddr =
      (unsigned int)(size_t)(__attribute__((address_space(3))) unsigned short*)lds_dst;
  u32x4g g0;
  g0[0] = 1u;                                             // count=1 (valid), user mode
  g0[1] = laddr;                                          // lds_addr (bytes)
  g0[2] = (unsigned int)ga;                               // global_addr[31:0]
  g0[3] = (unsigned int)((ga >> 32) & 0x01FFFFFFull)      // global_addr[56:32]
          | (2u << 30);                                   // type=2 ("image")
  const unsigned int d0 = 0x100000u, d1 = 0x100000u;      // generous OOB dims
  i32x8g g1;
  g1[0] = (int)(1u << 16);                                // data_size=1 -> 2 bytes
  g1[1] = (int)((d0 & 0xFFFFu) << 16);                    // tensor_dim0[15:0]
  g1[2] = (int)(((d0 >> 16) & 0xFFFFu) | ((d1 & 0xFFFFu) << 16));
  g1[3] = (int)(((d1 >> 16) & 0xFFFFu) | ((unsigned)tile_cols << 16)); // tile_dim0
  g1[4] = (int)(unsigned)tile_rows;                       // tile_dim1 (tile_dim2=0)
  g1[5] = row_stride_elems;                               // tensor_dim0_stride[31:0]
  g1[6] = 0;
  g1[7] = 0;
  i32x4g gz = {0, 0, 0, 0};
#if __clang_major__ >= 23
  i32x8g gz8 = {0, 0, 0, 0, 0, 0, 0, 0};
  __builtin_amdgcn_tensor_load_to_lds(g0, g1, gz, gz, gz8, 0);
#else
  __builtin_amdgcn_tensor_load_to_lds(g0, g1, gz, gz, 0);
#endif
}
#endif

__device__ __forceinline__ unsigned short f2bf(float f) {
  unsigned int u = __float_as_uint(f);
  u += 0x7FFFu + ((u >> 16) & 1u);          // round-to-nearest-even
  return (unsigned short)(u >> 16);
}

__device__ __forceinline__ v8f wmma_bf16(v16bf a, v16bf b, v8f c) {
  // D = A(16x32 bf16) * B(32x16 bf16) + C(16x16 f32)
  return __builtin_amdgcn_wmma_f32_16x16x32_bf16(
      /*neg_a=*/false, a, /*neg_b=*/false, b,
      /*c_mod=*/(short)0, c, /*reuse_a=*/false, /*reuse_b=*/false);
}

// ---------------------------------------------------------------------------
// Problem constants
// ---------------------------------------------------------------------------
#define BATCH 4
#define SEQ   4096
#define DIM   1024
#define HEADS 16
#define HD    64
#define MTOK  (BATCH * SEQ)     // 16384 tokens

// ---------------------------------------------------------------------------
// f32 -> bf16 conversion (row-major, vectorized)
// ---------------------------------------------------------------------------
__global__ __launch_bounds__(256)
void cvt_bf16_kernel(const float* __restrict__ in, unsigned short* __restrict__ out, int n4) {
  int i = blockIdx.x * 256 + threadIdx.x;
  if (i < n4) {
    f32x4 f = ((const f32x4*)in)[i];
    union { unsigned short u[4]; unsigned long long q; } r;
    r.u[0] = f2bf(f[0]); r.u[1] = f2bf(f[1]);
    r.u[2] = f2bf(f[2]); r.u[3] = f2bf(f[3]);
    ((unsigned long long*)out)[i] = r.q;
  }
}

// ---------------------------------------------------------------------------
// f32 (KxN) -> bf16 transposed (NxK): done ONCE so the GEMM inner loop needs
// no LDS transpose scatter and both tiles are contiguous 2D tiles (TDM-able).
// 32x32 tile through LDS, 32x8 threads.
// ---------------------------------------------------------------------------
__global__ __launch_bounds__(256)
void cvt_transpose_kernel(const float* __restrict__ in, unsigned short* __restrict__ out,
                          int K, int N) {
  __shared__ float tile[32][33];
  const int kb = blockIdx.y * 32, nb = blockIdx.x * 32;
  const int tx = threadIdx.x & 31, ty = threadIdx.x >> 5;
  #pragma unroll
  for (int i = 0; i < 4; ++i)
    tile[ty + 8 * i][tx] = in[(size_t)(kb + ty + 8 * i) * N + nb + tx];
  __syncthreads();
  #pragma unroll
  for (int i = 0; i < 4; ++i)
    out[(size_t)(nb + ty + 8 * i) * K + kb + tx] = f2bf(tile[tx][ty + 8 * i]);
}

// ---------------------------------------------------------------------------
// bf16 WMMA GEMM:  C(MxN f32) = A(MxK) * Bt(NxK)^T [+ bias]   (both row-major)
// Block tile 128x128x64, double-buffered LDS, ONE barrier per K-tile.
// 8 waves; wave computes 32(M) x 64(N) = 2x4 WMMA tiles, 16 WMMAs / K-tile.
// Tile fill: A via TENSOR_LOAD_TO_LDS (one D# per K-tile, TENSORcnt, wave 0),
// Bt via GLOBAL_LOAD_ASYNC_TO_LDS_B128 (ASYNCcnt), register fallbacks below.
// Fragment fetches are contiguous ds_load_b128 matching ISA 7.12.2 layouts:
//   A lane L: M=L%16, halves 0..7 -> K = 8*(L/16)..+7, halves 8..15 -> +16
//   B lane L: N=L%16, halves 0..7 -> K = 8*(L/16)..+7, halves 8..15 -> +16
// ---------------------------------------------------------------------------
#define BM  128
#define BN  128
#define BKO 64            // outer K tile (2 WMMA K-steps)

__global__ __launch_bounds__(256)
void gemm_bf16_kernel(const unsigned short* __restrict__ A,
                      const unsigned short* __restrict__ Bt,
                      float* __restrict__ C,
                      const float* __restrict__ bias,
                      int M, int N, int K) {
  __shared__ __align__(16) unsigned short sA[2][BM * BKO];    // 2 x 16 KB
  __shared__ __align__(16) unsigned short sBt[2][BN * BKO];   // 2 x 16 KB

  const int tid  = threadIdx.x;
  const int lane = tid & 31;
  const int wave = tid >> 5;
  const int wr   = wave >> 1;          // 0..3 : M offset wr*32
  const int wc   = wave & 1;           // 0..1 : N offset wc*64
  const int m0   = blockIdx.y * BM;
  const int n0   = blockIdx.x * BN;

  v8f zero = {};
  v8f acc[2][4];
  #pragma unroll
  for (int i = 0; i < 2; ++i)
    #pragma unroll
    for (int j = 0; j < 4; ++j) acc[i][j] = zero;

  // tile-copy thread mapping (same for A and Bt: 128 rows x 64 halves)
  const int trow = tid >> 1;            // 0..127
  const int tseg = (tid & 1) * 32;      // 0 / 32 halves of a 64-half row

  // fragment lane mapping
  const int m   = lane & 15;
  const int kk  = (lane >> 4) * 8;      // A fragment K-half base
  const int kk2 = (lane >> 4) * 16;     // unused (B fragment uses kk)

  const unsigned short* Abase  = A  + (size_t)(m0 + trow) * K + tseg;
  const unsigned short* Btbase = Bt + (size_t)(n0 + trow) * K + tseg;
  const int iters = K / BKO;
  int cur = 0;

#if !USE_TDM && !USE_ASYNC_LDS
  H32 ha;
#endif
#if !USE_ASYNC_LDS
  H32 hb;
#endif

  // ---- pipeline prologue: stage tile 0 ----
#if USE_TDM
  if (wave == 0)
    tdm_load_tile(A + (size_t)m0 * K, &sA[0][0], BKO, BM, K);
#elif USE_ASYNC_LDS
  #pragma unroll
  for (int t = 0; t < 4; ++t)
    async_b128(Abase + 8 * t, &sA[0][trow * BKO + tseg + 8 * t]);
#else
  #pragma unroll
  for (int t = 0; t < 4; ++t) ha.q[t] = ((const u32x4*)Abase)[t];
#endif
#if USE_ASYNC_LDS
  #pragma unroll
  for (int t = 0; t < 4; ++t)
    async_b128(Btbase + 8 * t, &sBt[0][trow * BKO + tseg + 8 * t]);
#else
  #pragma unroll
  for (int t = 0; t < 4; ++t) hb.q[t] = ((const u32x4*)Btbase)[t];
#endif

  for (int it = 0; it < iters; ++it) {
    const int kb = it * BKO;

    // ---- publish register-staged tiles (fallback paths only) ----
#if !USE_TDM && !USE_ASYNC_LDS
    {
      u32x4* da = (u32x4*)(&sA[cur][trow * BKO + tseg]);
      #pragma unroll
      for (int t = 0; t < 4; ++t) da[t] = ha.q[t];
    }
#endif
#if !USE_ASYNC_LDS
    {
      u32x4* db = (u32x4*)(&sBt[cur][trow * BKO + tseg]);
      #pragma unroll
      for (int t = 0; t < 4; ++t) db[t] = hb.q[t];
    }
#endif

    // ---- wait for DMA engines, then publish tile to the whole WGP ----
#if USE_TDM
    if (wave == 0) __builtin_amdgcn_s_wait_tensorcnt(0);
#endif
#if USE_ASYNC_LDS
    __builtin_amdgcn_s_wait_asynccnt(0);
#endif
    __syncthreads();

    // ---- issue next tile's fills (overlap with WMMA below) ----
    if (it + 1 < iters) {
      const unsigned short* An  = Abase  + kb + BKO;
      const unsigned short* Btn = Btbase + kb + BKO;
#if USE_TDM
      if (wave == 0)
        tdm_load_tile(A + (size_t)m0 * K + kb + BKO, &sA[cur ^ 1][0], BKO, BM, K);
#elif USE_ASYNC_LDS
      #pragma unroll
      for (int t = 0; t < 4; ++t)
        async_b128(An + 8 * t, &sA[cur ^ 1][trow * BKO + tseg + 8 * t]);
#else
      #pragma unroll
      for (int t = 0; t < 4; ++t) ha.q[t] = ((const u32x4*)An)[t];
#endif
#if USE_ASYNC_LDS
      #pragma unroll
      for (int t = 0; t < 4; ++t)
        async_b128(Btn + 8 * t, &sBt[cur ^ 1][trow * BKO + tseg + 8 * t]);
#else
      #pragma unroll
      for (int t = 0; t < 4; ++t) hb.q[t] = ((const u32x4*)Btn)[t];
#endif
    }
    if (it + 2 < iters)                          // warm L2 two tiles ahead
      __builtin_prefetch(Btbase + kb + 2 * BKO, 0, 2);

    // ---- 2 K-steps x 8 WMMAs from LDS buffer `cur` ----
    #pragma unroll
    for (int kc = 0; kc < 2; ++kc) {
      v16bf af[2], bf[4];
      #pragma unroll
      for (int mt = 0; mt < 2; ++mt) {
        Frag16 f;
        const unsigned short* p = &sA[cur][(wr * 32 + mt * 16 + m) * BKO + kc * 32];
        f.q[0] = *(const u32x4*)(p + kk);
        f.q[1] = *(const u32x4*)(p + 16 + kk);
        af[mt] = f.v;
      }
      #pragma unroll
      for (int nt = 0; nt < 4; ++nt) {
        Frag16 f;
        const unsigned short* p = &sBt[cur][(wc * 64 + nt * 16 + m) * BKO + kc * 32 + kk];
        f.q[0] = *(const u32x4*)(p);
        f.q[1] = *(const u32x4*)(p + 16);
        bf[nt] = f.v;
      }
      #pragma unroll
      for (int mt = 0; mt < 2; ++mt)
        #pragma unroll
        for (int nt = 0; nt < 4; ++nt)
          acc[mt][nt] = wmma_bf16(af[mt], bf[nt], acc[mt][nt]);
    }
    cur ^= 1;
  }

  // ---- epilogue: D element (M = r + 8*(lane/16), N = lane%16) ----
  const int rbase = (lane >> 4) * 8;
  #pragma unroll
  for (int mt = 0; mt < 2; ++mt) {
    #pragma unroll
    for (int nt = 0; nt < 4; ++nt) {
      const int c  = n0 + wc * 64 + nt * 16 + m;
      const float bv = bias ? bias[c] : 0.f;
      F8 d; d.v = acc[mt][nt];
      #pragma unroll
      for (int r = 0; r < 8; ++r) {
        const int row = m0 + wr * 32 + mt * 16 + rbase + r;
        C[(size_t)row * N + c] = d.f[r] + bv;
      }
    }
  }
}

// ---------------------------------------------------------------------------
// Per-token head-mixing attention: one wave per token.
// Lane layout for q/k/v: head = lane/2, 32 contiguous d-elements per lane.
// RMSNorm needs one shfl_xor(1); RoPE is lane-local on (re,im) pairs.
// S(16x16) = Q(16x64) K^T(64x16): 2x v_wmma_f32_16x16x32_bf16
// O(16x64) = P(16x16) V(16x64):   4x WMMA with K zero-padded 16->32
// ---------------------------------------------------------------------------
#define AW   8                       // tokens (waves) per block
#define SLAB (3 * HEADS * HD + HEADS * HEADS)   // 3328 ushorts / wave

__global__ __launch_bounds__(AW * 32)
void attn_head_mix_kernel(const float* __restrict__ qkv,
                          const float* __restrict__ cosp,
                          const float* __restrict__ sinp,
                          const float* __restrict__ gq,
                          const float* __restrict__ gk,
                          unsigned short* __restrict__ o) {
  __shared__ __align__(16) unsigned short lds[AW * SLAB];
  const int lane  = threadIdx.x & 31;
  const int w     = threadIdx.x >> 5;
  const int token = blockIdx.x * AW + w;

  unsigned short* qs = lds + w * SLAB;   // 16x64 bf16
  unsigned short* ks = qs + HEADS * HD;  // 16x64
  unsigned short* vs = ks + HEADS * HD;  // 16x64
  unsigned short* ps = vs + HEADS * HD;  // 16x16

  const int hd    = lane >> 1;
  const int half  = lane & 1;
  const int dbase = half * 32;

  // ---- load q,k,v (32 f32 each per lane, contiguous) ----
  const float* qp = qkv + (size_t)token * (3 * DIM) + hd * HD + dbase;
  float q[32], k[32], v[32];
  #pragma unroll
  for (int i = 0; i < 8; ++i) {
    f32x4 a = *(const f32x4*)(qp + 4 * i);
    f32x4 b = *(const f32x4*)(qp + DIM + 4 * i);
    f32x4 c = *(const f32x4*)(qp + 2 * DIM + 4 * i);
    #pragma unroll
    for (int j = 0; j < 4; ++j) { q[4*i+j] = a[j]; k[4*i+j] = b[j]; v[4*i+j] = c[j]; }
  }

  // ---- RMSNorm per head (d=64 spans 2 lanes) ----
  float ssq = 0.f, ssk = 0.f;
  #pragma unroll
  for (int i = 0; i < 32; ++i) { ssq += q[i]*q[i]; ssk += k[i]*k[i]; }
  ssq += __shfl_xor(ssq, 1, 32);
  ssk += __shfl_xor(ssk, 1, 32);
  const float invq = 1.f / (sqrtf(ssq) * 0.125f + 1e-6f);   // ||x||/sqrt(64)
  const float invk = 1.f / (sqrtf(ssk) * 0.125f + 1e-6f);
  #pragma unroll
  for (int i = 0; i < 8; ++i) {
    f32x4 ga = *(const f32x4*)(gq + dbase + 4 * i);
    f32x4 gb = *(const f32x4*)(gk + dbase + 4 * i);
    #pragma unroll
    for (int j = 0; j < 4; ++j) { q[4*i+j] *= invq * ga[j]; k[4*i+j] *= invk * gb[j]; }
  }

  // ---- RoPE (pairs are lane-local) ----
  const int nidx = token & (SEQ - 1);
  const float* cp = cosp + (size_t)nidx * (HD / 2) + half * 16;
  const float* sp = sinp + (size_t)nidx * (HD / 2) + half * 16;
  #pragma unroll
  for (int i = 0; i < 4; ++i) {
    f32x4 c4 = *(const f32x4*)(cp + 4 * i);
    f32x4 s4 = *(const f32x4*)(sp + 4 * i);
    #pragma unroll
    for (int j = 0; j < 4; ++j) {
      const int p = 4 * i + j;
      const float cr = c4[j], si = s4[j];
      float re = q[2*p], im = q[2*p+1];
      q[2*p] = re*cr - im*si;  q[2*p+1] = re*si + im*cr;
      re = k[2*p]; im = k[2*p+1];
      k[2*p] = re*cr - im*si;  k[2*p+1] = re*si + im*cr;
    }
  }

  // ---- stage bf16 tiles in LDS (vector b128 stores) ----
  H32 hq, hk, hv;
  #pragma unroll
  for (int i = 0; i < 32; ++i) { hq.u[i] = f2bf(q[i]); hk.u[i] = f2bf(k[i]); hv.u[i] = f2bf(v[i]); }
  {
    u32x4* dq = (u32x4*)(qs + hd * HD + dbase);
    u32x4* dk = (u32x4*)(ks + hd * HD + dbase);
    u32x4* dv = (u32x4*)(vs + hd * HD + dbase);
    #pragma unroll
    for (int t = 0; t < 4; ++t) { dq[t] = hq.q[t]; dk[t] = hk.q[t]; dv[t] = hv.q[t]; }
  }
  __syncthreads();

  // ---- S = Q K^T (16x16, K=64): two WMMAs ----
  const int m   = lane & 15;
  const int kkA = (lane >> 4) * 8;
  const int kkB = (lane >> 4) * 16;
  v16bf aq[2], bk[2];
  #pragma unroll
  for (int c = 0; c < 2; ++c) {
    Frag16 fa;                                   // A: row m of Q, K chunk c
    const unsigned short* pa = qs + m * HD + c * 32;
    fa.q[0] = *(const u32x4*)(pa + kkA);
    fa.q[1] = *(const u32x4*)(pa + 16 + kkA);
    aq[c] = fa.v;
    Frag16 fb;                                   // B: col m of K^T = row m of K
    const unsigned short* pb = ks + m * HD + c * 32 + kkA;
    fb.q[0] = *(const u32x4*)(pb);
    fb.q[1] = *(const u32x4*)(pb + 16);
    bk[c] = fb.v;
  }
  v8f zero = {};
  v8f s = wmma_bf16(aq[0], bk[0], zero);
  s = wmma_bf16(aq[1], bk[1], s);

  // ---- softmax over g (row M lives in one VGPR across a 16-lane half) ----
  F8 sf; sf.v = s;
  #pragma unroll
  for (int r = 0; r < 8; ++r) {
    float x = sf.f[r] * 0.125f;                  // scale = d^-0.5
    float mx = x;
    mx = fmaxf(mx, __shfl_xor(mx, 1, 16));
    mx = fmaxf(mx, __shfl_xor(mx, 2, 16));
    mx = fmaxf(mx, __shfl_xor(mx, 4, 16));
    mx = fmaxf(mx, __shfl_xor(mx, 8, 16));
    float e = __expf(x - mx);
    float sm = e;
    sm += __shfl_xor(sm, 1, 16);
    sm += __shfl_xor(sm, 2, 16);
    sm += __shfl_xor(sm, 4, 16);
    sm += __shfl_xor(sm, 8, 16);
    ps[(r + ((lane >> 4) << 3)) * 16 + m] = f2bf(e / sm);
  }
  __syncthreads();

  // ---- O = P V (K=16 zero-padded to 32): four WMMAs ----
  Frag16 fa;
  fa.q[0] = *(const u32x4*)(ps + m * 16 + kkA);  // K = kkA..kkA+7 (< 16)
  u32x4 zq = {};
  fa.q[1] = zq;                                  // K = 16..31 -> zero pad
  v16bf pmat = fa.v;
  #pragma unroll
  for (int j = 0; j < 4; ++j) {
    Frag16 fb;                                   // B: V rows g=0..15, cols j*16..+15
    #pragma unroll
    for (int jj = 0; jj < 16; ++jj) {
      unsigned short val = vs[(kkA + (jj & 7)) * HD + j * 16 + m];
      fb.u[jj] = (jj < 8) ? val : (unsigned short)0;   // elements 8-15: K>=16 pad
    }
    v8f ov = wmma_bf16(pmat, fb.v, zero);
    F8 of; of.v = ov;
    #pragma unroll
    for (int r = 0; r < 8; ++r) {
      const int h = r + ((lane >> 4) << 3);
      o[(size_t)token * DIM + h * HD + j * 16 + m] = f2bf(of.f[r]);
    }
  }
}

// ---------------------------------------------------------------------------
// Launch: cvt(x) + transpose(wqkv,wproj) -> GEMM qkv -> attention -> GEMM proj
// Workspace layout (bytes):
//   xb     bf16 [16384,1024]        @ 0          (32 MB)
//   wqkvbt bf16 [3072,1024] (N,K)   @ 33554432   ( 6 MB)
//   wprojbt bf16 [1024,1024] (N,K)  @ 39845888   ( 2 MB)
//   qkv    f32  [16384,3072]        @ 41943040   (192 MB)
//   ob     bf16 [16384,1024]        @ 243269632  (32 MB)
// ---------------------------------------------------------------------------
extern "C" void kernel_launch(void* const* d_in, const int* in_sizes, int n_in,
                              void* d_out, int out_size, void* d_ws, size_t ws_size,
                              hipStream_t stream) {
  (void)in_sizes; (void)n_in; (void)out_size; (void)ws_size;
  const float* x      = (const float*)d_in[0];
  const float* fcos   = (const float*)d_in[1];
  const float* fsin   = (const float*)d_in[2];
  const float* w_qkv  = (const float*)d_in[3];
  const float* g_q    = (const float*)d_in[4];
  const float* g_k    = (const float*)d_in[5];
  const float* w_proj = (const float*)d_in[6];
  const float* b_proj = (const float*)d_in[7];
  float* out = (float*)d_out;

  char* ws = (char*)d_ws;
  unsigned short* xb      = (unsigned short*)(ws + 0);
  unsigned short* wqkvbt  = (unsigned short*)(ws + 33554432ull);
  unsigned short* wprojbt = (unsigned short*)(ws + 39845888ull);
  float*          qkvf    = (float*)        (ws + 41943040ull);
  unsigned short* ob      = (unsigned short*)(ws + 243269632ull);

  // 1) f32 -> bf16 conversions (weights transposed to N-major, once)
  {
    int n4 = MTOK * DIM / 4;
    cvt_bf16_kernel<<<(n4 + 255) / 256, 256, 0, stream>>>(x, xb, n4);
    cvt_transpose_kernel<<<dim3(3 * DIM / 32, DIM / 32), 256, 0, stream>>>(
        w_qkv, wqkvbt, DIM, 3 * DIM);
    cvt_transpose_kernel<<<dim3(DIM / 32, DIM / 32), 256, 0, stream>>>(
        w_proj, wprojbt, DIM, DIM);
  }

  // 2) qkv = x @ w_qkv   (16384 x 3072, K=1024)
  gemm_bf16_kernel<<<dim3(3 * DIM / BN, MTOK / BM), 256, 0, stream>>>(
      xb, wqkvbt, qkvf, nullptr, MTOK, 3 * DIM, DIM);

  // 3) per-token head-mix attention (RMSNorm + RoPE + softmax(QK^T)V)
  attn_head_mix_kernel<<<MTOK / AW, AW * 32, 0, stream>>>(
      qkvf, fcos, fsin, g_q, g_k, ob);

  // 4) out = o @ w_proj + b_proj   (16384 x 1024, K=1024)
  gemm_bf16_kernel<<<dim3(DIM / BN, MTOK / BM), 256, 0, stream>>>(
      ob, wprojbt, out, b_proj, MTOK, DIM, DIM);
}
